// StyleGAN2GeneratorFull_45028437131965
// MI455X (gfx1250) — matrix operand order repaired and checked by reading, not compiled, not run on hardware
//
#include <hip/hip_runtime.h>
#include <math.h>
#include <stdint.h>

// ---------------------------------------------------------------------------
// Types for CDNA5 WMMA (gfx1250, wave32)
// ---------------------------------------------------------------------------
typedef __attribute__((ext_vector_type(16))) __bf16 v16bf;
typedef __attribute__((ext_vector_type(8)))  __bf16 v8bf;
typedef __attribute__((ext_vector_type(8)))  float  v8f;
typedef __attribute__((ext_vector_type(4)))  unsigned int u32x4;
typedef __attribute__((ext_vector_type(8)))  int i32x8;
typedef __attribute__((ext_vector_type(4)))  int i32x4;

#define SQRT2F 1.41421356237309515f

// ---------------------------------------------------------------------------
// Mapping network: PixelNorm + 8x EqualLinear(lr_mul=0.01) + fused lrelu
// ---------------------------------------------------------------------------
struct MapP { const float* w[8]; const float* b[8]; };

__global__ __launch_bounds__(512)
void mapping_kernel(const float* __restrict__ styles, MapP mp, float* __restrict__ latent)
{
    __shared__ float xs[4][512];
    __shared__ float xn[4][512];
    __shared__ float rn[4];
    const int tid = threadIdx.x;

    for (int b = 0; b < 4; ++b) xs[b][tid] = styles[b * 512 + tid];
    __syncthreads();

    if (tid < 4) {
        float ss = 0.f;
        for (int k = 0; k < 512; ++k) { float v = xs[tid][k]; ss += v * v; }
        rn[tid] = rsqrtf(ss * (1.f / 512.f) + 1e-8f);
    }
    __syncthreads();
    for (int b = 0; b < 4; ++b) xs[b][tid] *= rn[b];
    __syncthreads();

    const float scale = 0.01f / sqrtf(512.f);
    for (int l = 0; l < 8; ++l) {
        const float* Wm = mp.w[l];
        const float* Bm = mp.b[l];
        float acc0 = 0.f, acc1 = 0.f, acc2 = 0.f, acc3 = 0.f;
        for (int k = 0; k < 512; ++k) {
            float wv = Wm[tid * 512 + k];
            acc0 += xs[0][k] * wv; acc1 += xs[1][k] * wv;
            acc2 += xs[2][k] * wv; acc3 += xs[3][k] * wv;
        }
        float bias = Bm[tid] * 0.01f;
        float a[4] = { acc0, acc1, acc2, acc3 };
        for (int b = 0; b < 4; ++b) {
            float v = a[b] * scale + bias;
            v = (v > 0.f ? v : 0.2f * v) * SQRT2F;
            xn[b][tid] = v;
        }
        __syncthreads();
        for (int b = 0; b < 4; ++b) xs[b][tid] = xn[b][tid];
        __syncthreads();
    }
    for (int b = 0; b < 4; ++b) latent[b * 512 + tid] = xs[b][tid];
}

// s[b,ci] = latent[b,:] . mod_w[ci,:]/sqrt(512) + mod_b[ci]
__global__ void style_kernel(const float* __restrict__ latent,
                             const float* __restrict__ mw,
                             const float* __restrict__ mb,
                             float* __restrict__ s, int Cin)
{
    int idx = blockIdx.x * blockDim.x + threadIdx.x;
    if (idx >= 4 * Cin) return;
    int b = idx / Cin, c = idx - b * Cin;
    float acc = 0.f;
    const float* lp = latent + b * 512;
    const float* wp = mw + (size_t)c * 512;
    for (int k = 0; k < 512; ++k) acc += lp[k] * wp[k];
    s[idx] = acc * 0.044194173824159216f + mb[c];
}

// wsq[co,ci] = scale^2 * sum_tap w[co,ci,tap]^2
__global__ void wsq_kernel(const float* __restrict__ w, float* __restrict__ wsq,
                           int Cout, int Cin, float scale2)
{
    int idx = blockIdx.x * blockDim.x + threadIdx.x;
    if (idx >= Cout * Cin) return;
    const float* p = w + (size_t)idx * 9;
    float a = 0.f;
#pragma unroll
    for (int t = 0; t < 9; ++t) a += p[t] * p[t];
    wsq[idx] = a * scale2;
}

// d[b,co] = rsqrt( sum_ci wsq[co,ci]*s[b,ci]^2 + 1e-8 )
__global__ void demod_kernel(const float* __restrict__ s, const float* __restrict__ wsq,
                             float* __restrict__ dv, int Cout, int Cin)
{
    int idx = blockIdx.x * blockDim.x + threadIdx.x;
    if (idx >= 4 * Cout) return;
    int b = idx / Cout, co = idx - b * Cout;
    const float* sp = s + b * Cin;
    const float* wp = wsq + (size_t)co * Cin;
    float a = 0.f;
    for (int ci = 0; ci < Cin; ++ci) { float sv = sp[ci]; a += wp[ci] * sv * sv; }
    dv[idx] = rsqrtf(a + 1e-8f);
}

// ---------------------------------------------------------------------------
// Weight prep: f32 (Cout,Cin,3,3) -> bf16 tap-major (Cout, tap, Cin), scaled.
// ---------------------------------------------------------------------------
__global__ void wprep_kernel(const float* __restrict__ w, __bf16* __restrict__ wb,
                             int Cout, int Cin, float scale)
{
    int idx = blockIdx.x * blockDim.x + threadIdx.x;
    int total = Cout * 9 * Cin;
    if (idx >= total) return;
    int ci  = idx % Cin;
    int r   = idx / Cin;
    int tap = r % 9;
    int co  = r / 9;
    wb[idx] = (__bf16)(scale * w[((size_t)co * Cin + ci) * 9 + tap]);
}

// ---------------------------------------------------------------------------
// TDM warm: issue a Tensor Data Mover 2D tile load of the prepped weights
// into LDS (D# per CDNA5 ISA ch.8), wait TENSORcnt. Pure L2 prefetch —
// no observable state. This toolchain exposes the 6-arg builtin
// (g0 u32x4, g1 i32x8, g2 i32x4, g3 i32x4, i32x8, cpol).
// ---------------------------------------------------------------------------
__global__ __launch_bounds__(32)
void tdm_warm_kernel(const __bf16* wb, int Cout, int K, float* sink)
{
#if defined(__gfx1250__) && __has_builtin(__builtin_amdgcn_tensor_load_to_lds)
    __shared__ __align__(128) __bf16 stage[64 * 64];
    unsigned long long ga = (unsigned long long)(uintptr_t)wb;

    u32x4 g0;
    g0[0] = 1u;                                        // count=1, user mode
    g0[1] = 0u;                                        // lds_addr = 0
    g0[2] = (unsigned)(ga & 0xffffffffu);              // global_addr[31:0]
    g0[3] = (unsigned)((ga >> 32) & 0x01ffffffu)       // global_addr[56:32]
          | (2u << 30);                                // type = 2 ("image")

    unsigned td0 = (unsigned)K;      // tensor_dim0 (elements)
    unsigned td1 = (unsigned)Cout;   // tensor_dim1 (rows)
    unsigned tl0 = 64u, tl1 = 64u;   // tile dims
    unsigned long long st0 = (unsigned long long)K;    // dim0 stride (elements)

    i32x8 g1;
    g1[0] = (int)(1u << 16);                               // data_size=1 -> 2 bytes
    g1[1] = (int)((td0 & 0xffffu) << 16);                  // tensor_dim0 lo16
    g1[2] = (int)((td0 >> 16) | ((td1 & 0xffffu) << 16));  // dim0 hi16 | dim1 lo16
    g1[3] = (int)((td1 >> 16) | (tl0 << 16));              // dim1 hi16 | tile_dim0
    g1[4] = (int)tl1;                                      // tile_dim1 (tile_dim2=0)
    g1[5] = (int)(st0 & 0xffffffffu);                      // dim0_stride[31:0]
    g1[6] = (int)((st0 >> 32) & 0xffffu);                  // dim0_stride[47:32]
    g1[7] = 0;
    i32x4 g2 = {};
    i32x4 g3 = {};
    i32x8 g4 = {};

    __builtin_amdgcn_tensor_load_to_lds(g0, g1, g2, g3, g4, 0);
#if __has_builtin(__builtin_amdgcn_s_wait_tensorcnt)
    __builtin_amdgcn_s_wait_tensorcnt(0);
#endif
    // keep `stage` alive; never true at runtime
    float v = (float)stage[threadIdx.x];
    if (v > 1.0e30f) sink[threadIdx.x] = v;
#else
    (void)wb; (void)Cout; (void)K; (void)sink;
#endif
}

// ---------------------------------------------------------------------------
// Modulate + transpose: x (b,ci,p) f32 -> xm (b,p,ci) bf16, x *= s[b,ci].
// ---------------------------------------------------------------------------
__global__ __launch_bounds__(256)
void xmod_kernel(const float* __restrict__ x, const float* __restrict__ s,
                 __bf16* __restrict__ xm, int Cin, int HW)
{
    __shared__ float tile[32][33];
    const int b  = blockIdx.z;
    const int c0 = blockIdx.y * 32;
    const int p0 = blockIdx.x * 32;
    const int tx = threadIdx.x & 31;
    const int ty = threadIdx.x >> 5;
#pragma unroll
    for (int r = 0; r < 4; ++r) {
        int ci = c0 + ty + r * 8;
        int p  = p0 + tx;
        float v = 0.f;
        if (p < HW) v = x[((size_t)b * Cin + ci) * HW + p] * s[b * Cin + ci];
        tile[ty + r * 8][tx] = v;
    }
    __syncthreads();
#pragma unroll
    for (int r = 0; r < 4; ++r) {
        int p  = p0 + ty + r * 8;
        int ci = c0 + tx;
        if (p < HW)
            xm[((size_t)b * HW + p) * Cin + ci] = (__bf16)tile[tx][ty + r * 8];
    }
}

// ---------------------------------------------------------------------------
// WMMA implicit-GEMM 3x3 conv, tap-major K, bf16 pre-transformed operands,
// register-level software pipelining of global staging.
// Block: 256 thr = 8 wave32, tile M=128 x N=128, K-step 32, 8 WMMA/wave/step.
// ---------------------------------------------------------------------------
#define LDA 40   // padded LDS row stride (bf16): 80B, conflict-free, 16B-aligned
__global__ __launch_bounds__(256)
void conv3x3_wmma(const __bf16* __restrict__ xm, const __bf16* __restrict__ wb,
                  const float* __restrict__ dvec, float* __restrict__ out,
                  int Cin, int Cout, int Hin, int Win, int Hout, int Wout,
                  int upsample)
{
    __shared__ __align__(16) __bf16 As[128 * LDA];
    __shared__ __align__(16) __bf16 Bs[128 * LDA];

    const int K    = Cin * 9;
    const int HW   = Hout * Wout;
    const int Ntot = 4 * HW;
    const int tid  = threadIdx.x;
    const int lane = tid & 31;
    const int wave = tid >> 5;
    const int wm   = wave & 3;
    const int wn   = wave >> 2;
    const int n0b  = blockIdx.x * 128;
    const int m0b  = blockIdx.y * 128;

    v8f acc[2][4] = {};

    const int row2 = tid >> 1;          // 0..127: A row / B col staged by this thread
    const int seg  = (tid & 1) * 16;    // 16-elem K segment

    const int npix  = n0b + row2;
    const bool nvalid = npix < Ntot;
    int pb = 0, oy = 0, ox = 0;
    if (nvalid) { pb = npix / HW; int r = npix - pb * HW; oy = r / Wout; ox = r - oy * Wout; }

    const __bf16* arow_p = wb + (size_t)(m0b + row2) * K;   // contiguous A row

    v8bf ra0, ra1, rb0, rb1;   // pipelined staging registers

    auto stage_load = [&](int kc) {
        const int tap = kc / Cin;           // uniform per step (Cin % 32 == 0)
        const int cib = kc - tap * Cin;
        const int ky = tap / 3, kx = tap - ky * 3;
        const __bf16* asrc = arow_p + kc + seg;
        if (kc + 32 < K) __builtin_prefetch(asrc + 32, 0, 1);  // global_prefetch_b8
        ra0 = *(const v8bf*)asrc;
        ra1 = *(const v8bf*)(asrc + 8);
        bool valid = nvalid;
        int iy = 0, ix = 0;
        if (!upsample) {
            iy = oy + ky - 1; ix = ox + kx - 1;
            valid = valid && iy >= 0 && iy < Hin && ix >= 0 && ix < Win;
        } else {
            int u = oy - ky, v = ox - kx;
            valid = valid && u >= 0 && v >= 0 && !(u & 1) && !(v & 1);
            iy = u >> 1; ix = v >> 1;
            valid = valid && iy < Hin && ix < Win;
        }
        if (valid) {
            const __bf16* bsrc = xm + ((size_t)(pb * Hin + iy) * Win + ix) * Cin + cib + seg;
            rb0 = *(const v8bf*)bsrc;
            rb1 = *(const v8bf*)(bsrc + 8);
        } else {
            v8bf z = {};
            rb0 = z; rb1 = z;
        }
    };

    stage_load(0);

    for (int k0 = 0; k0 < K; k0 += 32) {
        __syncthreads();   // previous compute done reading LDS
        *(v8bf*)(As + row2 * LDA + seg)     = ra0;
        *(v8bf*)(As + row2 * LDA + seg + 8) = ra1;
        *(v8bf*)(Bs + row2 * LDA + seg)     = rb0;
        *(v8bf*)(Bs + row2 * LDA + seg + 8) = rb1;
        __syncthreads();   // tiles visible to all waves
        if (k0 + 32 < K) stage_load(k0 + 32);   // overlaps with compute below

        const int khalf = lane >> 4;
        const int rsel  = lane & 15;
        v16bf av[2];
#pragma unroll
        for (int ms = 0; ms < 2; ++ms) {
            const __bf16* ap = As + (wm * 32 + ms * 16 + rsel) * LDA + khalf * 8;
            v8bf a0 = *(const v8bf*)ap;
            v8bf a1 = *(const v8bf*)(ap + 16);
            av[ms] = __builtin_shufflevector(a0, a1,
                        0,1,2,3,4,5,6,7,8,9,10,11,12,13,14,15);
        }
#pragma unroll
        for (int t = 0; t < 4; ++t) {
            const __bf16* bp = Bs + (wn * 64 + t * 16 + rsel) * LDA + khalf * 8;
            v8bf b0 = *(const v8bf*)bp;
            v8bf b1 = *(const v8bf*)(bp + 16);
            v16bf bv = __builtin_shufflevector(b0, b1,
                        0,1,2,3,4,5,6,7,8,9,10,11,12,13,14,15);
#pragma unroll
            for (int ms = 0; ms < 2; ++ms)
                acc[ms][t] = __builtin_amdgcn_wmma_f32_16x16x32_bf16(
                                 false, av[ms], false, bv, (short)0, acc[ms][t], false, false);
        }
    }

    // epilogue: demodulate by d[b,co], scatter to NCHW
    const int nloc  = lane & 15;
    const int mhalf = (lane >> 4) * 8;
#pragma unroll
    for (int t = 0; t < 4; ++t) {
        int n = n0b + wn * 64 + t * 16 + nloc;
        if (n >= Ntot) continue;
        int b = n / HW; int r = n - b * HW;
        int yy = r / Wout; int xx = r - yy * Wout;
#pragma unroll
        for (int ms = 0; ms < 2; ++ms) {
            int mbase = m0b + wm * 32 + ms * 16 + mhalf;
#pragma unroll
            for (int i = 0; i < 8; ++i) {
                int co = mbase + i;
                float v = acc[ms][t][i] * dvec[b * Cout + co];
                out[(((size_t)b * Cout + co) * Hout + yy) * Wout + xx] = v;
            }
        }
    }
}

// ---------------------------------------------------------------------------
// Depthwise 4x4 blur (outer([1,3,3,1])/16), pad 1, (2H+1) -> 2H
// ---------------------------------------------------------------------------
__global__ void blur_kernel(const float* __restrict__ t, float* __restrict__ y,
                            int C, int Hin, int Hout)
{
    int idx = blockIdx.x * blockDim.x + threadIdx.x;
    int total = 4 * C * Hout * Hout;
    if (idx >= total) return;
    int bc = idx / (Hout * Hout);
    int p  = idx - bc * Hout * Hout;
    int oy = p / Hout, ox = p - oy * Hout;
    const float cf[4] = { 1.f, 3.f, 3.f, 1.f };
    const float* src = t + (size_t)bc * Hin * Hin;
    float acc = 0.f;
#pragma unroll
    for (int i = 0; i < 4; ++i) {
        int iy = oy - 1 + i;
        if (iy < 0 || iy >= Hin) continue;
#pragma unroll
        for (int j = 0; j < 4; ++j) {
            int ix = ox - 1 + j;
            if (ix < 0 || ix >= Hin) continue;
            acc += cf[i] * cf[j] * src[iy * Hin + ix];
        }
    }
    y[idx] = acc * (1.f / 16.f);
}

// y = lrelu(y + bias[c] + nw*noise[h,w]) * sqrt(2), in-place
__global__ void epilogue_kernel(float* __restrict__ y, const float* __restrict__ bias,
                                const float* __restrict__ noise, const float* __restrict__ nw,
                                int C, int H)
{
    int idx = blockIdx.x * blockDim.x + threadIdx.x;
    int total = 4 * C * H * H;
    if (idx >= total) return;
    int c = (idx / (H * H)) % C;
    int p = idx % (H * H);
    float v = y[idx] + bias[c] + nw[0] * noise[p];
    y[idx] = (v > 0.f ? v : 0.2f * v) * SQRT2F;
}

// to_rgb: 1x1 modulated conv (no demod) + bias; out (4,3,H,H)
__global__ void rgb_kernel(const float* __restrict__ x, const float* __restrict__ w,
                           const float* __restrict__ s, const float* __restrict__ bias,
                           float* __restrict__ outp, int Cin, int H)
{
    int idx = blockIdx.x * blockDim.x + threadIdx.x;
    int HW = H * H;
    if (idx >= 4 * HW) return;
    int b = idx / HW; int p = idx - b * HW;
    float scale = rsqrtf((float)Cin);
    float a0 = 0.f, a1 = 0.f, a2 = 0.f;
    for (int ci = 0; ci < Cin; ++ci) {
        float coeff = s[b * Cin + ci] * scale * x[((size_t)(b * Cin + ci)) * HW + p];
        a0 += w[ci] * coeff;
        a1 += w[Cin + ci] * coeff;
        a2 += w[2 * Cin + ci] * coeff;
    }
    outp[((size_t)(b * 3 + 0)) * HW + p] = a0 + bias[0];
    outp[((size_t)(b * 3 + 1)) * HW + p] = a1 + bias[1];
    outp[((size_t)(b * 3 + 2)) * HW + p] = a2 + bias[2];
}

// out = rgb + bilinear_upsample_2x(skip)  (half-pixel centers, edge clamp)
__global__ void skip_up_add_kernel(const float* __restrict__ rgb,
                                   const float* __restrict__ skip,
                                   float* __restrict__ outp, int Hs, int Ho)
{
    int idx = blockIdx.x * blockDim.x + threadIdx.x;
    int total = 4 * 3 * Ho * Ho;
    if (idx >= total) return;
    int bc = idx / (Ho * Ho);
    int p  = idx - bc * Ho * Ho;
    int oy = p / Ho, ox = p - oy * Ho;
    float fy = oy * 0.5f - 0.25f;
    float fx = ox * 0.5f - 0.25f;
    int y0 = (int)floorf(fy), x0 = (int)floorf(fx);
    float wy = fy - (float)y0, wx = fx - (float)x0;
    int y1 = y0 + 1, x1 = x0 + 1;
    y0 = min(max(y0, 0), Hs - 1); y1 = min(max(y1, 0), Hs - 1);
    x0 = min(max(x0, 0), Hs - 1); x1 = min(max(x1, 0), Hs - 1);
    const float* sp = skip + (size_t)bc * Hs * Hs;
    float v = (1.f - wy) * ((1.f - wx) * sp[y0 * Hs + x0] + wx * sp[y0 * Hs + x1])
            +        wy  * ((1.f - wx) * sp[y1 * Hs + x0] + wx * sp[y1 * Hs + x1]);
    outp[idx] = rgb[idx] + v;
}

// broadcast const (1,512,4,4) -> (4,512,4,4)
__global__ void bcast_kernel(const float* __restrict__ c, float* __restrict__ outp, int per)
{
    int idx = blockIdx.x * blockDim.x + threadIdx.x;
    if (idx >= 4 * per) return;
    outp[idx] = c[idx % per];
}

// ---------------------------------------------------------------------------
// Host orchestration. Flat input order (depth-first, dict insertion order):
//   0 styles; 1..16 mapping (w,b)x8; 17 const;
//   18..22 conv1 {w, mod.w, mod.b, noise_w, bias};
//   23..26 to_rgb1 {w, mod.w, mod.b, bias};
//   27..86 convs[12] x {w, mod.w, mod.b, noise_w, bias};
//   87..110 to_rgbs[6] x {w, mod.w, mod.b, bias};
//   111..123 noises[13]
// ---------------------------------------------------------------------------
extern "C" void kernel_launch(void* const* d_in, const int* in_sizes, int n_in,
                              void* d_out, int out_size, void* d_ws, size_t ws_size,
                              hipStream_t stream)
{
    (void)in_sizes; (void)n_in; (void)out_size; (void)ws_size;
    auto di = [&](int i) { return (const float*)d_in[i]; };
    float* ws = (float*)d_ws;

    // workspace layout (float elements; every block 16B-aligned)
    size_t o = 0;
    float* latent = ws + o; o += 2048;
    float* svec   = ws + o; o += 2048;
    float* wsqb   = ws + o; o += 512 * 512;
    float* dvec   = ws + o; o += 2048;
    float* tdms   = ws + o; o += 64;          // TDM warm sink (never written)
    float* rgbT   = ws + o; o += 786432;
    float* skipA  = ws + o; o += 786432;
    float* skipB  = ws + o; o += 786432;
    __bf16* wbf   = (__bf16*)(ws + o); o += 1179648;   // <= 512*9*512 bf16
    __bf16* xmb   = (__bf16*)(ws + o); o += 17000000;  // <= 4*256*256*128 bf16
    const size_t BIGN = 33900544;  // >= 4*128*257*257
    float* Wb[3] = { ws + o, ws + o + BIGN, ws + o + 2 * BIGN };

    // ---- mapping network -> latent
    MapP mp;
    for (int l = 0; l < 8; ++l) { mp.w[l] = di(1 + 2 * l); mp.b[l] = di(2 + 2 * l); }
    mapping_kernel<<<1, 512, 0, stream>>>(di(0), mp, latent);

    // ---- const input broadcast -> Wb[0]
    bcast_kernel<<<(4 * 8192 + 255) / 256, 256, 0, stream>>>(di(17), Wb[0], 8192);

    auto styled_conv_raw = [&](const float* xin, float* outp, int Cin, int Cout,
                               int Hin, int Hout, int up,
                               const float* wp, const float* mw, const float* mb) {
        style_kernel<<<(4 * Cin + 255) / 256, 256, 0, stream>>>(latent, mw, mb, svec, Cin);
        wsq_kernel<<<(Cout * Cin + 255) / 256, 256, 0, stream>>>(wp, wsqb, Cout, Cin,
                                                                 1.f / ((float)Cin * 9.f));
        demod_kernel<<<(4 * Cout + 255) / 256, 256, 0, stream>>>(svec, wsqb, dvec, Cout, Cin);
        float scale = 1.f / sqrtf((float)Cin * 9.f);
        wprep_kernel<<<(Cout * 9 * Cin + 255) / 256, 256, 0, stream>>>(wp, wbf, Cout, Cin, scale);
        int HWin = Hin * Hin;
        dim3 gx((HWin + 31) / 32, Cin / 32, 4);
        xmod_kernel<<<gx, 256, 0, stream>>>(xin, svec, xmb, Cin, HWin);
        tdm_warm_kernel<<<1, 32, 0, stream>>>(wbf, Cout, Cin * 9, tdms);
        int Ntot = 4 * Hout * Hout;
        dim3 grid((Ntot + 127) / 128, Cout / 128);
        conv3x3_wmma<<<grid, 256, 0, stream>>>(xmb, wbf, dvec, outp,
                                               Cin, Cout, Hin, Hin, Hout, Hout, up);
    };

    // ---- conv1 @ 4x4 (no upsample)
    styled_conv_raw(Wb[0], Wb[1], 512, 512, 4, 4, 0, di(18), di(19), di(20));
    epilogue_kernel<<<(4 * 512 * 16 + 255) / 256, 256, 0, stream>>>(
        Wb[1], di(22), di(111), di(21), 512, 4);

    // ---- to_rgb1 -> skipA
    style_kernel<<<(4 * 512 + 255) / 256, 256, 0, stream>>>(latent, di(24), di(25), svec, 512);
    rgb_kernel<<<(4 * 16 + 255) / 256, 256, 0, stream>>>(Wb[1], di(23), svec, di(26),
                                                         skipA, 512, 4);

    int cur = 1;
    int in_c = 512, H = 4;
    float* skip  = skipA;
    float* skipO = skipB;
    const int outc_tab[6] = { 512, 512, 512, 512, 256, 128 };

    for (int j = 0; j < 6; ++j) {
        int out_c = outc_tab[j];
        int cA = 27 + 10 * j;      // convs[2j]   (upsample)
        int cB = cA + 5;           // convs[2j+1]
        int rb = 87 + 4 * j;       // to_rgbs[j]
        int ni = 111 + 1 + 2 * j;  // noise index for conv A
        float* xb = Wb[cur];
        float* tb = Wb[(cur + 1) % 3];

        // upsample conv: (4,in_c,H,H) -> raw (4,out_c,2H+1,2H+1)
        styled_conv_raw(xb, tb, in_c, out_c, H, 2 * H + 1, 1, di(cA), di(cA + 1), di(cA + 2));
        int elems = 4 * out_c * 4 * H * H;
        blur_kernel<<<(elems + 255) / 256, 256, 0, stream>>>(tb, xb, out_c, 2 * H + 1, 2 * H);
        epilogue_kernel<<<(elems + 255) / 256, 256, 0, stream>>>(
            xb, di(cA + 4), di(ni), di(cA + 3), out_c, 2 * H);

        // second conv (same res): xb -> tb
        styled_conv_raw(xb, tb, out_c, out_c, 2 * H, 2 * H, 0, di(cB), di(cB + 1), di(cB + 2));
        epilogue_kernel<<<(elems + 255) / 256, 256, 0, stream>>>(
            tb, di(cB + 4), di(ni + 1), di(cB + 3), out_c, 2 * H);

        // to_rgb + skip upsample-add
        style_kernel<<<(4 * out_c + 255) / 256, 256, 0, stream>>>(latent, di(rb + 1), di(rb + 2),
                                                                  svec, out_c);
        int pix = 4 * 4 * H * H;
        rgb_kernel<<<(pix + 255) / 256, 256, 0, stream>>>(tb, di(rb), svec, di(rb + 3),
                                                          rgbT, out_c, 2 * H);
        float* dst = (j == 5) ? (float*)d_out : skipO;
        skip_up_add_kernel<<<(3 * pix + 255) / 256, 256, 0, stream>>>(rgbT, skip, dst, H, 2 * H);
        { float* old = skip; skip = dst; skipO = old; }

        cur = (cur + 1) % 3;
        in_c = out_c;
        H *= 2;
    }
}
